// OnehotEncoder_75634374082604
// MI455X (gfx1250) — compile-verified, matrix operand order checked by reading, not run on hardware
//
#include <hip/hip_runtime.h>
#include <hip/hip_bf16.h>
#include <math.h>

// ---------------------------------------------------------------------------
// GRU encoder, T=64, B=1024, V=1024, E=512, fp32 in/out.
//
// Strategy (MI455X / gfx1250, wave32, WMMA bf16):
//   1. Fold W_comb = W_ih @ W_d2e  [1536,1024], b_comb = W_ih@b_d2e + b_ih.
//   2. One big batched GEMM: gi_all[T*B,1536] = X[T*B,1024] @ W_comb^T + b_comb
//      using v_wmma_f32_16x16x32_bf16 (f32 accumulate), double-buffered LDS.
//   3. 64 sequential fused step kernels: gh = h @ W_hh^T (WMMA bf16),
//      fused sigmoid/tanh GRU update, ping-pong h buffers (f32 + bf16 mirror).
// Workspace: ~395 MB (gi_all 384 MB dominates).
// ---------------------------------------------------------------------------

typedef __bf16 bf16_t;
typedef bf16_t v16bf __attribute__((ext_vector_type(16)));
typedef bf16_t v8bf  __attribute__((ext_vector_type(8)));
typedef float  v8f   __attribute__((ext_vector_type(8)));
typedef float  v4f   __attribute__((ext_vector_type(4)));

#define T_STEPS 64
#define B_SZ    1024
#define V_SZ    1024
#define E_SZ    512
#define G3E     1536          // 3*E
#define LDA     40            // LDS row pitch in bf16 (80B: 16B aligned, bank-skewed)

// Load one 16x32 (A) or 32x16-as-[N][K] (B) bf16 WMMA fragment from an LDS
// tile stored row-major [16 rows][32 k] with pitch LDA.
// wave32 layout: lanes 0-15 row/col = lane, K = {0..7,16..23};
//                lanes 16-31 same row/col, K = {8..15,24..31}.
__device__ __forceinline__ v16bf load_frag(const bf16_t* base, int lane) {
    const int r  = lane & 15;
    const int kh = (lane >> 4) << 3;              // 0 or 8
    const bf16_t* p = base + r * LDA;
    v8bf lo = *(const v8bf*)(p + kh);             // K = kh .. kh+7
    v8bf hi = *(const v8bf*)(p + 16 + kh);        // K = 16+kh .. 16+kh+7
    v16bf f;
#pragma unroll
    for (int i = 0; i < 8; ++i) { f[i] = lo[i]; f[i + 8] = hi[i]; }
    return f;
}

__device__ __forceinline__ float sigmoidf_fast(float x) {
    return 1.0f / (1.0f + __expf(-x));
}

// ---------------------------------------------------------------------------
// Prep kernels
// ---------------------------------------------------------------------------

// W_comb[n,k] = sum_j W_ih[n,j] * W_d2e[j,k]   (n<1536, j<512, k<1024) -> bf16
__global__ __launch_bounds__(256)
void fold_w_kernel(const float* __restrict__ W_ih, const float* __restrict__ W_d2e,
                   bf16_t* __restrict__ Wc) {
    const int k = blockIdx.x * blockDim.x + threadIdx.x;   // 0..1023
    const int n = blockIdx.y;                              // 0..1535 (uniform -> s_loads)
    float s = 0.f;
    for (int j = 0; j < E_SZ; ++j)
        s = fmaf(W_ih[(size_t)n * E_SZ + j], W_d2e[(size_t)j * V_SZ + k], s);
    Wc[(size_t)n * V_SZ + k] = (bf16_t)s;
}

// b_comb[n] = b_ih[n] + sum_j W_ih[n,j] * b_d2e[j]
__global__ __launch_bounds__(256)
void fold_b_kernel(const float* __restrict__ W_ih, const float* __restrict__ b_d2e,
                   const float* __restrict__ b_ih, float* __restrict__ bc) {
    const int n = blockIdx.x * blockDim.x + threadIdx.x;
    if (n >= G3E) return;
    float s = b_ih[n];
    for (int j = 0; j < E_SZ; ++j)
        s = fmaf(W_ih[(size_t)n * E_SZ + j], b_d2e[j], s);
    bc[n] = s;
}

// Convert W_hh -> bf16 and zero the initial hidden-state buffers.
__global__ __launch_bounds__(256)
void init_kernel(const float* __restrict__ W_hh, bf16_t* __restrict__ Whh_bf,
                 float* __restrict__ h0f, bf16_t* __restrict__ h0b) {
    const int i = blockIdx.x * blockDim.x + threadIdx.x;
    if (i < G3E * E_SZ) Whh_bf[i] = (bf16_t)W_hh[i];
    if (i < B_SZ * E_SZ) { h0f[i] = 0.f; h0b[i] = (bf16_t)0.f; }
}

// ---------------------------------------------------------------------------
// Register-staged tile copies for the big GEMM (256 threads/block).
// A: 128 rows x 32 k of f32 X -> bf16 LDS.  B: 128 rows x 32 k of bf16 Wc.
// All global loads issue back-to-back (one loadcnt wait), stores later.
// ---------------------------------------------------------------------------
__device__ __forceinline__ void g1_load_a(const float* __restrict__ X, int rowBase,
                                          int kk, int tid, v4f a[4]) {
    const int r0 = tid >> 3, kg = (tid & 7) * 4;
#pragma unroll
    for (int i = 0; i < 4; ++i)
        a[i] = *(const v4f*)(X + (size_t)(rowBase + r0 + i * 32) * V_SZ + kk + kg);
}
__device__ __forceinline__ void g1_store_a(bf16_t* Als, int tid, const v4f a[4]) {
    const int r0 = tid >> 3, kg = (tid & 7) * 4;
#pragma unroll
    for (int i = 0; i < 4; ++i) {
        bf16_t* d = &Als[(r0 + i * 32) * LDA + kg];
        d[0] = (bf16_t)a[i][0]; d[1] = (bf16_t)a[i][1];
        d[2] = (bf16_t)a[i][2]; d[3] = (bf16_t)a[i][3];
    }
}
__device__ __forceinline__ void g1_load_b(const bf16_t* __restrict__ Wc, int colBase,
                                          int kk, int tid, v8bf b[2]) {
    const int n0 = tid >> 2, kg = (tid & 3) * 8;
#pragma unroll
    for (int i = 0; i < 2; ++i)
        b[i] = *(const v8bf*)(Wc + (size_t)(colBase + n0 + i * 64) * V_SZ + kk + kg);
}
__device__ __forceinline__ void g1_store_b(bf16_t* Bls, int tid, const v8bf b[2]) {
    const int n0 = tid >> 2, kg = (tid & 3) * 8;
#pragma unroll
    for (int i = 0; i < 2; ++i)
        *(v8bf*)&Bls[(n0 + i * 64) * LDA + kg] = b[i];
}

// ---------------------------------------------------------------------------
// Big batched GEMM: gi[M=65536, N=1536] = X[M,1024](f32->bf16) @ Wc^T + bc
// Block: 256 threads = 8 waves, tile 128x128, K-chunk 32, double-buffered LDS.
// Wave grid 2(M) x 4(N): each wave 64x32 -> 4x2 wmma accumulators.
// ---------------------------------------------------------------------------
__global__ __launch_bounds__(256)
void gi_gemm_kernel(const float* __restrict__ X, const bf16_t* __restrict__ Wc,
                    const float* __restrict__ bc, float* __restrict__ gi) {
    __shared__ bf16_t Als[2][128 * LDA];
    __shared__ bf16_t Bls[2][128 * LDA];

    const int tid  = threadIdx.x;
    const int lane = tid & 31;
    const int wave = tid >> 5;
    const int wm   = wave & 1;          // 0..1 -> 64-row group
    const int wn   = wave >> 1;         // 0..3 -> 32-col group
    const int rowBase = blockIdx.x * 128;
    const int colBase = blockIdx.y * 128;

    v8f acc[4][2] = {};

    // Prologue: stage K-chunk 0 into buffer 0.
    {
        v4f a0[4]; v8bf b0[2];
        g1_load_a(X, rowBase, 0, tid, a0);
        g1_load_b(Wc, colBase, 0, tid, b0);
        g1_store_a(&Als[0][0], tid, a0);
        g1_store_b(&Bls[0][0], tid, b0);
    }

    for (int kk = 0; kk < V_SZ; kk += 32) {
        const int  cur  = (kk >> 5) & 1;
        const bool more = (kk + 32) < V_SZ;
        v4f anext[4]; v8bf bnext[2];
        if (more) {                       // issue next chunk's global loads early
            g1_load_a(X, rowBase, kk + 32, tid, anext);
            g1_load_b(Wc, colBase, kk + 32, tid, bnext);
        }
        __syncthreads();                  // buffer `cur` fully staged

        v16bf af[4], bfr[2];
#pragma unroll
        for (int mi = 0; mi < 4; ++mi)
            af[mi] = load_frag(&Als[cur][(wm * 64 + mi * 16) * LDA], lane);
#pragma unroll
        for (int ni = 0; ni < 2; ++ni)
            bfr[ni] = load_frag(&Bls[cur][(wn * 32 + ni * 16) * LDA], lane);
#pragma unroll
        for (int mi = 0; mi < 4; ++mi)
#pragma unroll
            for (int ni = 0; ni < 2; ++ni)
                acc[mi][ni] = __builtin_amdgcn_wmma_f32_16x16x32_bf16(
                    false, af[mi], false, bfr[ni], (short)0, acc[mi][ni],
                    false, false);

        if (more) {                       // drain loads into the other buffer
            g1_store_a(&Als[cur ^ 1][0], tid, anext);
            g1_store_b(&Bls[cur ^ 1][0], tid, bnext);
        }
    }

    // Epilogue: D layout -> row M = r + 8*(lane>=16), col N = lane&15.
    const int hi8   = (lane >> 4) * 8;
    const int nLane = lane & 15;
#pragma unroll
    for (int mi = 0; mi < 4; ++mi) {
#pragma unroll
        for (int ni = 0; ni < 2; ++ni) {
            const int col = colBase + wn * 32 + ni * 16 + nLane;
            const float bias = bc[col];
#pragma unroll
            for (int r = 0; r < 8; ++r) {
                const int row = rowBase + wm * 64 + mi * 16 + hi8 + r;
                gi[(size_t)row * G3E + col] = acc[mi][ni][r] + bias;
            }
        }
    }
}

// ---------------------------------------------------------------------------
// Register-staged tile copies for the step kernel (128 threads/block).
// A: 64 rows x 32 k of bf16 h.  B: 3 gates x 32 cols x 32 k of bf16 W_hh.
// ---------------------------------------------------------------------------
__device__ __forceinline__ void st_load_a(const bf16_t* __restrict__ h, int rowBase,
                                          int kk, int tid, v8bf a[2]) {
    const int r0 = tid >> 2, kg = (tid & 3) * 8;
#pragma unroll
    for (int i = 0; i < 2; ++i)
        a[i] = *(const v8bf*)(h + (size_t)(rowBase + r0 + i * 32) * E_SZ + kk + kg);
}
__device__ __forceinline__ void st_store_a(bf16_t* Als, int tid, const v8bf a[2]) {
    const int r0 = tid >> 2, kg = (tid & 3) * 8;
#pragma unroll
    for (int i = 0; i < 2; ++i)
        *(v8bf*)&Als[(r0 + i * 32) * LDA + kg] = a[i];
}
__device__ __forceinline__ void st_load_b(const bf16_t* __restrict__ Whh, int eBase,
                                          int kk, int tid, v8bf b[3]) {
    const int nl = tid >> 2, kg = (tid & 3) * 8;       // nl < 32 -> gate == i
#pragma unroll
    for (int i = 0; i < 3; ++i)
        b[i] = *(const v8bf*)(Whh + (size_t)(i * E_SZ + eBase + nl) * E_SZ + kk + kg);
}
__device__ __forceinline__ void st_store_b(bf16_t* Bls, int tid, const v8bf b[3]) {
    const int nl = tid >> 2, kg = (tid & 3) * 8;
#pragma unroll
    for (int i = 0; i < 3; ++i)
        *(v8bf*)&Bls[(nl + i * 32) * LDA + kg] = b[i];
}

// ---------------------------------------------------------------------------
// Fused GRU step: gh = h @ W_hh^T (WMMA), then gate math, write h_new.
// Block: 128 threads = 4 waves; tile 64 rows x 32 e-cols x 3 gates,
// double-buffered LDS. Each wave: 16 rows; acc[3 gates][2 n-frags]. The three
// gate values for one (row,e) land in the same lane/register -> fusion free.
// ---------------------------------------------------------------------------
__global__ __launch_bounds__(128)
void gru_step_kernel(const bf16_t* __restrict__ hin_bf,  // [B,E] bf16
                     const float*  __restrict__ hin_f,   // [B,E] f32
                     const bf16_t* __restrict__ Whh,     // [3E,E] bf16
                     const float*  __restrict__ b_hh,    // [3E]
                     const float*  __restrict__ gi_t,    // [B,3E] slice at t
                     float*        __restrict__ hout_f,  // [B,E]
                     bf16_t*       __restrict__ hout_bf) // [B,E]
{
    __shared__ bf16_t Als[2][64 * LDA];   // 64 rows x 32 k
    __shared__ bf16_t Bls[2][96 * LDA];   // 3 gates x 32 cols x 32 k

    const int tid  = threadIdx.x;
    const int lane = tid & 31;
    const int wave = tid >> 5;                 // 0..3 -> 16-row group
    const int rowBase = blockIdx.x * 64;
    const int eBase   = blockIdx.y * 32;

    v8f acc[3][2] = {};

    {
        v8bf a0[2], b0[3];
        st_load_a(hin_bf, rowBase, 0, tid, a0);
        st_load_b(Whh, eBase, 0, tid, b0);
        st_store_a(&Als[0][0], tid, a0);
        st_store_b(&Bls[0][0], tid, b0);
    }

    for (int kk = 0; kk < E_SZ; kk += 32) {
        const int  cur  = (kk >> 5) & 1;
        const bool more = (kk + 32) < E_SZ;
        v8bf anext[2], bnext[3];
        if (more) {
            st_load_a(hin_bf, rowBase, kk + 32, tid, anext);
            st_load_b(Whh, eBase, kk + 32, tid, bnext);
        }
        __syncthreads();

        v16bf a = load_frag(&Als[cur][(wave * 16) * LDA], lane);
#pragma unroll
        for (int g = 0; g < 3; ++g)
#pragma unroll
            for (int ni = 0; ni < 2; ++ni) {
                v16bf b = load_frag(&Bls[cur][(g * 32 + ni * 16) * LDA], lane);
                acc[g][ni] = __builtin_amdgcn_wmma_f32_16x16x32_bf16(
                    false, a, false, b, (short)0, acc[g][ni], false, false);
            }

        if (more) {
            st_store_a(&Als[cur ^ 1][0], tid, anext);
            st_store_b(&Bls[cur ^ 1][0], tid, bnext);
        }
    }

    const int hi8   = (lane >> 4) * 8;
    const int nLane = lane & 15;
#pragma unroll
    for (int ni = 0; ni < 2; ++ni) {
        const int e = eBase + ni * 16 + nLane;
        const float bh_r = b_hh[e];
        const float bh_z = b_hh[E_SZ + e];
        const float bh_n = b_hh[2 * E_SZ + e];
#pragma unroll
        for (int r8 = 0; r8 < 8; ++r8) {
            const int row = rowBase + wave * 16 + hi8 + r8;
            const size_t gbase = (size_t)row * G3E;
            const float gh_r = acc[0][ni][r8] + bh_r;
            const float gh_z = acc[1][ni][r8] + bh_z;
            const float gh_n = acc[2][ni][r8] + bh_n;
            const float gi_r = gi_t[gbase + e];
            const float gi_z = gi_t[gbase + E_SZ + e];
            const float gi_n = gi_t[gbase + 2 * E_SZ + e];
            const float rg = sigmoidf_fast(gi_r + gh_r);
            const float zg = sigmoidf_fast(gi_z + gh_z);
            const float ng = tanhf(gi_n + rg * gh_n);
            const float hold = hin_f[(size_t)row * E_SZ + e];
            const float hnew = (1.0f - zg) * ng + zg * hold;
            hout_f [(size_t)row * E_SZ + e] = hnew;
            hout_bf[(size_t)row * E_SZ + e] = (bf16_t)hnew;
        }
    }
}

// ---------------------------------------------------------------------------
extern "C" void kernel_launch(void* const* d_in, const int* in_sizes, int n_in,
                              void* d_out, int out_size, void* d_ws, size_t ws_size,
                              hipStream_t stream) {
    (void)in_sizes; (void)n_in; (void)out_size; (void)ws_size;

    const float* X     = (const float*)d_in[0];   // [T,B,V]
    const float* W_d2e = (const float*)d_in[1];   // [E,V]
    const float* b_d2e = (const float*)d_in[2];   // [E]
    const float* W_ih  = (const float*)d_in[3];   // [3E,E]
    const float* W_hh  = (const float*)d_in[4];   // [3E,E]
    const float* b_ih  = (const float*)d_in[5];   // [3E]
    const float* b_hh  = (const float*)d_in[6];   // [3E]
    float* out = (float*)d_out;                   // [B,E]

    // Workspace carve-up (all chunks are multiples of 256 B).
    char* ws = (char*)d_ws;
    size_t off = 0;
    float*  gi     = (float*) (ws + off); off += (size_t)T_STEPS * B_SZ * G3E * 4; // 384 MB
    bf16_t* Wc     = (bf16_t*)(ws + off); off += (size_t)G3E * V_SZ * 2;           // 3 MB
    float*  bc     = (float*) (ws + off); off += (size_t)G3E * 4;
    bf16_t* Whh_bf = (bf16_t*)(ws + off); off += (size_t)G3E * E_SZ * 2;
    float*  hf0    = (float*) (ws + off); off += (size_t)B_SZ * E_SZ * 4;
    float*  hf1    = (float*) (ws + off); off += (size_t)B_SZ * E_SZ * 4;
    bf16_t* hb0    = (bf16_t*)(ws + off); off += (size_t)B_SZ * E_SZ * 2;
    bf16_t* hb1    = (bf16_t*)(ws + off); off += (size_t)B_SZ * E_SZ * 2;

    float*  hf[2] = { hf0, hf1 };
    bf16_t* hb[2] = { hb0, hb1 };

    // 1. Init: W_hh -> bf16, h0 = 0.
    init_kernel<<<(G3E * E_SZ + 255) / 256, 256, 0, stream>>>(W_hh, Whh_bf, hf[0], hb[0]);
    // 2. Fold input path: W_comb = W_ih @ W_d2e, b_comb = W_ih @ b_d2e + b_ih.
    fold_w_kernel<<<dim3(V_SZ / 256, G3E), 256, 0, stream>>>(W_ih, W_d2e, Wc);
    fold_b_kernel<<<(G3E + 255) / 256, 256, 0, stream>>>(W_ih, b_d2e, b_ih, bc);
    // 3. All input-side gates in one batched WMMA GEMM.
    gi_gemm_kernel<<<dim3((T_STEPS * B_SZ) / 128, G3E / 128), 256, 0, stream>>>(X, Wc, bc, gi);
    // 4. Sequential fused recurrence (ping-pong h buffers).
    for (int t = 0; t < T_STEPS; ++t) {
        const int cur = t & 1, nxt = cur ^ 1;
        const float* gi_t = gi + (size_t)t * B_SZ * G3E;
        float* hof = (t == T_STEPS - 1) ? out : hf[nxt];
        gru_step_kernel<<<dim3(B_SZ / 64, E_SZ / 32), 128, 0, stream>>>(
            hb[cur], hf[cur], Whh_bf, b_hh, gi_t, hof, hb[nxt]);
    }
}
